// MoEGate_15728170238345
// MI455X (gfx1250) — matrix operand order repaired and checked by reading, not compiled, not run on hardware
//
#include <hip/hip_runtime.h>
#include <hip/hip_bf16.h>

#define HIDDEN 5120
#define NEXP   160
#define NTILES 10            // 160 / 16
#define TOPKG  3
#define TOPK   6

#define ROW_BYTES    320     // 160 experts * 2B bf16
#define LDS_STRIDE   336     // padded row stride (21 * 16B) -> 16B aligned, low bank conflicts
#define TILE_BYTES   (32 * LDS_STRIDE)          // one 32-K B panel: 10752 B
#define CHUNKS       ((32 * ROW_BYTES) / 16)    // 640 x 16B chunks of real data

typedef __attribute__((ext_vector_type(16))) __bf16 v16bf;
typedef __attribute__((ext_vector_type(8)))  __bf16 v8bf;
typedef __attribute__((ext_vector_type(8)))  float  v8f;

// exact pointee type expected by the async builtin (per hipcc diagnostic):
// "__attribute__((__vector_size__(4 * sizeof(int)))) int __device__ *"
typedef int v4i_gcc __attribute__((vector_size(16)));
typedef __attribute__((address_space(1))) v4i_gcc GV4;   // global (AS1)
typedef __attribute__((address_space(3))) v4i_gcc LV4;   // LDS    (AS3)

#if __has_builtin(__builtin_amdgcn_global_load_async_to_lds_b128)
#define HAVE_ASYNC_LDS 1
#endif

__device__ __forceinline__ void async_b128(const void* g, void* l) {
#ifdef HAVE_ASYNC_LDS
    __builtin_amdgcn_global_load_async_to_lds_b128(
        (GV4*)(unsigned long long)(uintptr_t)g,
        (LV4*)(unsigned int)(uintptr_t)l,      // flat->LDS offset (addr[31:0])
        0, 0);
#else
    *(float4*)l = *(const float4*)g;            // fallback: through VGPRs
#endif
}

__device__ __forceinline__ void async_drain() {
#ifdef HAVE_ASYNC_LDS
#if __has_builtin(__builtin_amdgcn_s_wait_asynccnt)
    __builtin_amdgcn_s_wait_asynccnt(0);
#else
    asm volatile("s_wait_asynccnt 0x0" ::: "memory");
#endif
#endif
}

// ---------------------------------------------------------------------------
// Kernel 1: W [160,5120] f32 (row-major) -> Wt [5120,160] bf16 (K-major)
// ---------------------------------------------------------------------------
__global__ void convert_w_kernel(const float* __restrict__ W,
                                 __bf16* __restrict__ Wt) {
    int i = blockIdx.x * blockDim.x + threadIdx.x;
    if (i >= NEXP * HIDDEN) return;
    int k = i / NEXP;
    int n = i - k * NEXP;
    Wt[i] = (__bf16)W[(size_t)n * HIDDEN + k];
}

// ---------------------------------------------------------------------------
// Kernel 2: logits[T,160] = X[T,5120] * Wt  via v_wmma_f32_16x16x32_bf16.
// Block = 8 waves = 128 tokens. The shared 32x160 B panel is async-DMA'd
// into LDS (double buffered) once per block per K-step and reused by all
// 8 waves -> 8x less L2 traffic for Wt. B fragments are software-pipelined
// one N-tile ahead so WMMA execution hides the ds_load latency.
// ---------------------------------------------------------------------------
__global__ void __launch_bounds__(256)
gemm_logits_kernel(const float* __restrict__ X,
                   const __bf16* __restrict__ Wt,
                   float* __restrict__ logits, int T) {
    __shared__ __attribute__((aligned(16))) char smem[2 * TILE_BYTES];

    const int tid     = threadIdx.x;
    const int lane    = tid & 31;
    const int tile_m  = (blockIdx.x * 8 + (tid >> 5)) * 16;   // may exceed T
    const int p       = lane & 15;   // A row in tile / C column N
    const int h       = lane >> 4;   // half-wave selector

    // ---- stage B panel for kb=0 into buffer 0 ----
    for (int c = tid; c < CHUNKS; c += 256) {
        int row = c / 20, col = c - row * 20;
        async_b128((const char*)Wt + (size_t)row * ROW_BYTES + col * 16,
                   smem + row * LDS_STRIDE + col * 16);
    }
    async_drain();
    __syncthreads();

    v8f zero = {};
    v8f acc[NTILES];
#pragma unroll
    for (int nt = 0; nt < NTILES; ++nt) acc[nt] = zero;

    int rowA = tile_m + p; if (rowA >= T) rowA = T - 1;       // clamp, no return
    const float* xrow = X + (size_t)rowA * HIDDEN;

    for (int kb = 0; kb < HIDDEN; kb += 32) {
        const int bufOff = ((kb >> 5) & 1) * TILE_BYTES;

        // ---- async-prefetch next B panel into the other buffer ----
        if (kb + 32 < HIDDEN) {
            const int nxtOff = bufOff ^ TILE_BYTES;
            for (int c = tid; c < CHUNKS; c += 256) {
                int row = c / 20, col = c - row * 20;
                async_b128((const char*)Wt + (size_t)(kb + 32 + row) * ROW_BYTES + col * 16,
                           smem + nxtOff + row * LDS_STRIDE + col * 16);
            }
        }

        // ---- A fragment: ISA 16-bit A 16x32 layout ----
        // lanes 0-15 : K = kb+0..7  and kb+16..23  (row M = p)
        // lanes 16-31: K = kb+8..15 and kb+24..31  (row M = p)
        const float* alo = xrow + kb + h * 8;
        const float* ahi = alo + 16;
        float4 f0 = *(const float4*)(alo);
        float4 f1 = *(const float4*)(alo + 4);
        float4 f2 = *(const float4*)(ahi);
        float4 f3 = *(const float4*)(ahi + 4);

        if (kb + 32 < HIDDEN)
            __builtin_prefetch(xrow + kb + 32 + h * 8, 0, 3);

        v16bf a;
        a[0]  = (__bf16)f0.x;  a[1]  = (__bf16)f0.y;
        a[2]  = (__bf16)f0.z;  a[3]  = (__bf16)f0.w;
        a[4]  = (__bf16)f1.x;  a[5]  = (__bf16)f1.y;
        a[6]  = (__bf16)f1.z;  a[7]  = (__bf16)f1.w;
        a[8]  = (__bf16)f2.x;  a[9]  = (__bf16)f2.y;
        a[10] = (__bf16)f2.z;  a[11] = (__bf16)f2.w;
        a[12] = (__bf16)f3.x;  a[13] = (__bf16)f3.y;
        a[14] = (__bf16)f3.z;  a[15] = (__bf16)f3.w;

        // ---- B fragments from LDS: lane holds B column N = nt*16+p, K = kb+8h+0..7 and kb+16+8h+0..7 ----
        // Software-pipelined one tile ahead: load frag nt+1 before WMMA nt so
        // the compiler can use a partial s_wait_dscnt and hide LDS latency.
        const char* brow = smem + bufOff + h * 8 * LDS_STRIDE + p * 2;
        v8bf blo, bhi;
#pragma unroll
        for (int i = 0; i < 8; ++i) {
            blo[i] = *(const __bf16*)(brow + i * LDS_STRIDE);
            bhi[i] = *(const __bf16*)(brow + (16 + i) * LDS_STRIDE);
        }
#pragma unroll
        for (int nt = 0; nt < NTILES; ++nt) {
            v8bf nlo = blo, nhi = bhi;
            if (nt + 1 < NTILES) {
#pragma unroll
                for (int i = 0; i < 8; ++i) {
                    nlo[i] = *(const __bf16*)(brow + i * LDS_STRIDE + (nt + 1) * 32);
                    nhi[i] = *(const __bf16*)(brow + (16 + i) * LDS_STRIDE + (nt + 1) * 32);
                }
            }
            v16bf b;
#pragma unroll
            for (int i = 0; i < 8; ++i) { b[i] = blo[i]; b[8 + i] = bhi[i]; }
            acc[nt] = __builtin_amdgcn_wmma_f32_16x16x32_bf16(
                false, a, false, b, (short)0, acc[nt], false, false);
            blo = nlo; bhi = nhi;
        }

        async_drain();      // this wave's panel DMA complete
        __syncthreads();    // all waves: done reading bufOff, done staging other buf
    }

    // ---- store: C/D layout lane -> N = p ; VGPR r -> M = r + 8*h ----
#pragma unroll
    for (int nt = 0; nt < NTILES; ++nt) {
#pragma unroll
        for (int r = 0; r < 8; ++r) {
            int row = tile_m + r + 8 * h;
            if (row < T)
                logits[(size_t)row * NEXP + nt * 16 + p] = acc[nt][r];
        }
    }
}

// ---------------------------------------------------------------------------
// Kernel 3: softmax + group-limited top-k gating. One wave per token,
// 5 consecutive experts per lane (group of 20 == 4 lanes).
// ---------------------------------------------------------------------------
__global__ void __launch_bounds__(256)
gate_topk_kernel(const float* __restrict__ logits,
                 float* __restrict__ out_idx,
                 float* __restrict__ out_w, int T) {
    const int lane = threadIdx.x & 31;
    const int t    = (blockIdx.x * blockDim.x + threadIdx.x) >> 5;
    if (t >= T) return;

    const float* lrow = logits + (size_t)t * NEXP;
    float s[5];
#pragma unroll
    for (int j = 0; j < 5; ++j) s[j] = lrow[lane * 5 + j];

    // ---- softmax over 160 ----
    float m = s[0];
#pragma unroll
    for (int j = 1; j < 5; ++j) m = fmaxf(m, s[j]);
#pragma unroll
    for (int off = 16; off >= 1; off >>= 1) m = fmaxf(m, __shfl_xor(m, off, 32));
    float sum = 0.0f;
#pragma unroll
    for (int j = 0; j < 5; ++j) { s[j] = __expf(s[j] - m); sum += s[j]; }
#pragma unroll
    for (int off = 16; off >= 1; off >>= 1) sum += __shfl_xor(sum, off, 32);
    float inv = 1.0f / sum;
#pragma unroll
    for (int j = 0; j < 5; ++j) s[j] *= inv;

    // ---- group max (group g = lane/4 covers experts 20g..20g+19) ----
    float gm = s[0];
#pragma unroll
    for (int j = 1; j < 5; ++j) gm = fmaxf(gm, s[j]);
    gm = fmaxf(gm, __shfl_xor(gm, 1, 32));
    gm = fmaxf(gm, __shfl_xor(gm, 2, 32));

    float gv[8];
#pragma unroll
    for (int g = 0; g < 8; ++g) gv[g] = __shfl(gm, g * 4, 32);

    const int myg = lane >> 2;
    int rank = 0;
#pragma unroll
    for (int g = 0; g < 8; ++g)
        rank += (gv[g] > gv[myg]) || (gv[g] == gv[myg] && g < myg);
    if (rank >= TOPKG) {
#pragma unroll
        for (int j = 0; j < 5; ++j) s[j] = 0.0f;   // mask non-top groups
    }

    // ---- top-6 (value desc, index asc on ties, like jax.lax.top_k) ----
    float wk[TOPK]; int wi[TOPK];
#pragma unroll
    for (int k = 0; k < TOPK; ++k) {
        float bv = s[0]; int bj = 0;
#pragma unroll
        for (int j = 1; j < 5; ++j) if (s[j] > bv) { bv = s[j]; bj = j; }
        int bi = lane * 5 + bj;
#pragma unroll
        for (int off = 16; off >= 1; off >>= 1) {
            float ov = __shfl_xor(bv, off, 32);
            int   oi = __shfl_xor(bi, off, 32);
            if (ov > bv || (ov == bv && oi < bi)) { bv = ov; bi = oi; }
        }
        wk[k] = bv; wi[k] = bi;
        int loc = bi - lane * 5;
        if (loc >= 0 && loc < 5) s[loc] = -1.0f;   // eliminate picked expert
    }

    float tsum = 1e-20f;
#pragma unroll
    for (int k = 0; k < TOPK; ++k) tsum += wk[k];
    float tinv = 1.0f / tsum;

    if (lane == 0) {
#pragma unroll
        for (int k = 0; k < TOPK; ++k) {
            out_idx[(size_t)t * TOPK + k] = (float)wi[k];   // indices, value-cast
            out_w  [(size_t)t * TOPK + k] = wk[k] * tinv;
        }
    }
}

// ---------------------------------------------------------------------------
extern "C" void kernel_launch(void* const* d_in, const int* in_sizes, int n_in,
                              void* d_out, int out_size, void* d_ws, size_t ws_size,
                              hipStream_t stream) {
    const float* x = (const float*)d_in[0];   // [T, 5120] f32
    const float* w = (const float*)d_in[1];   // [160, 5120] f32
    const int T = in_sizes[0] / HIDDEN;       // 16384

    // workspace layout: [ Wt bf16 (1.6MB) | pad | logits f32 (T*160) ]
    __bf16* wt = (__bf16*)d_ws;
    size_t  wt_bytes = ((size_t)NEXP * HIDDEN * sizeof(__bf16) + 255) & ~(size_t)255;
    float*  logits = (float*)((char*)d_ws + wt_bytes);

    float* out_idx = (float*)d_out;               // first T*6: expert indices
    float* out_w   = out_idx + (size_t)T * TOPK;  // next  T*6: normalized weights

    // 1) convert + transpose router weights to bf16
    convert_w_kernel<<<(NEXP * HIDDEN + 255) / 256, 256, 0, stream>>>(w, wt);

    // 2) WMMA GEMM: 8 waves / 128 tokens per block, LDS-staged B panel
    int blocks = (T + 127) / 128;
    gemm_logits_kernel<<<blocks, 256, 0, stream>>>(x, wt, logits, T);

    // 3) gating: one wave per token
    int gblocks = (T * 32 + 255) / 256;
    gate_topk_kernel<<<gblocks, 256, 0, stream>>>(logits, out_idx, out_w, T);
}
